// HR2O_NL_4552665334425
// MI455X (gfx1250) — matrix operand order repaired and checked by reading, not compiled, not run on hardware
//
#include <hip/hip_runtime.h>
#include <hip/hip_bf16.h>

typedef __attribute__((ext_vector_type(16))) _Float16 v16h;
typedef __attribute__((ext_vector_type(8)))  _Float16 v8h;
typedef __attribute__((ext_vector_type(8)))  float    v8f;

#define NIMG 32
#define CCH  512
#define HH   32
#define WW   32
#define HW   (HH*WW)

union Frag { v16h v; v8h h[2]; };

static __device__ __forceinline__ v8f wmma_f16(const Frag& A, const Frag& B, v8f c) {
    return __builtin_amdgcn_wmma_f32_16x16x32_f16(false, A.v, false, B.v, (short)0, c, false, false);
}

// ---------------------------------------------------------------------------
// Repack conv weights: w[co][ci][kh][kw] f32  ->  wt[t][co][ci] f16 (t=kh*3+kw)
// ---------------------------------------------------------------------------
__global__ __launch_bounds__(256) void repack_w_kernel(const float* __restrict__ w,
                                                       _Float16* __restrict__ wt) {
    int idx = blockIdx.x * 256 + threadIdx.x;           // [t][co][ci]
    if (idx >= 9 * CCH * CCH) return;
    int ci = idx & 511;
    int co = (idx >> 9) & 511;
    int t  = idx >> 18;
    wt[idx] = (_Float16)w[(co * CCH + ci) * 9 + t];
}

// ---------------------------------------------------------------------------
// 3x3 SAME conv, 512->512, implicit GEMM with WMMA f16->f32.
// Block tile: 4 output rows (M = 4x32 w-positions) x 64 output channels.
// 8 waves = 4 row-tiles x 2 co-pairs; each wave computes a 32x32 tile
// (acc[2][2]) so 8 ds_load_b128 feed 4 WMMAs per tap.
// mode 0: out_h[hw][n][co] (f16).   mode 1: out_f (NCHW f32) = resid + conv.
// ---------------------------------------------------------------------------
__global__ __launch_bounds__(256) void conv3x3_wmma_kernel(
    const float* __restrict__ xin,          // NCHW f32
    const _Float16* __restrict__ wt,        // [9][512][512] f16
    _Float16* __restrict__ out_h,           // [hw][n][c] f16 (mode 0)
    float* __restrict__ out_f,              // NCHW f32 (mode 1)
    const float* __restrict__ resid,        // NCHW f32 (mode 1)
    int mode)
{
    __shared__ _Float16 xs[6 * 34 * 32];    // [row(4+halo)][col(+halo)][ci] 13.1 KB
    __shared__ _Float16 ws[9 * 64 * 32];    // [tap][co][ci]                36.9 KB

    const int tid  = threadIdx.x;
    const int wave = tid >> 5;
    const int lane = tid & 31;
    const int g    = lane >> 4;             // lane half-group
    const int ln   = lane & 15;             // m (A-row) or n (B-col) in tile
    const int mrow = wave >> 1;             // 0..3 : which output row of the block
    const int nt2  = wave & 1;              // 0..1 : 32-wide co pair

    const int nimg = blockIdx.x >> 3;       // image
    const int h0   = (blockIdx.x & 7) * 4;  // first output row of block
    const int co0  = blockIdx.y * 64;       // out-channel group

    const float* xbase = xin + (size_t)nimg * CCH * HW;

    v8f acc[2][2] = {};

    for (int cc = 0; cc < CCH / 32; ++cc) {
        __syncthreads();
        // ---- stage x halo slab: xs[r][col][ci] = x[n, cc*32+ci, h0-1+r, col-1]
        for (int e = tid; e < 6 * 34 * 32; e += 256) {
            int col = e % 34;
            int ci  = (e / 34) & 31;
            int r   = e / (34 * 32);
            int gy  = h0 - 1 + r;
            int gx  = col - 1;
            float val = 0.0f;
            if (gy >= 0 && gy < HH && gx >= 0 && gx < WW)
                val = xbase[((cc * 32 + ci) * HH + gy) * WW + gx];
            xs[(r * 34 + col) * 32 + ci] = (_Float16)val;
        }
        // ---- stage weight tile: ws[t][co][ci] (vectorized, 8 halves/thread)
        for (int vix = tid; vix < (9 * 64 * 32) / 8; vix += 256) {
            int ci8 = (vix & 3) * 8;
            int co  = (vix >> 2) & 63;
            int t   = vix >> 8;
            *(v8h*)&ws[(t * 64 + co) * 32 + ci8] =
                *(const v8h*)&wt[((size_t)(t * CCH + co0 + co)) * CCH + cc * 32 + ci8];
        }
        __syncthreads();

        // ---- 9 taps x (2x2) 16x16x32 WMMA with shared fragments
        #pragma unroll
        for (int t = 0; t < 9; ++t) {
            const int dh = t / 3, dw = t % 3;
            const int rA = mrow + dh;                  // 0..5
            Frag A0, A1, B0, B1;
            {
                const int colA = ln + dw;              // mi = 0
                A0.h[0] = *(const v8h*)&xs[(rA * 34 + colA) * 32 + g * 8];
                A0.h[1] = *(const v8h*)&xs[(rA * 34 + colA) * 32 + 16 + g * 8];
            }
            {
                const int colA = 16 + ln + dw;         // mi = 1
                A1.h[0] = *(const v8h*)&xs[(rA * 34 + colA) * 32 + g * 8];
                A1.h[1] = *(const v8h*)&xs[(rA * 34 + colA) * 32 + 16 + g * 8];
            }
            {
                const _Float16* wp = &ws[(t * 64 + nt2 * 32 + ln) * 32 + g * 8];
                B0.h[0] = *(const v8h*)(wp);
                B0.h[1] = *(const v8h*)(wp + 16);
            }
            {
                const _Float16* wp = &ws[(t * 64 + nt2 * 32 + 16 + ln) * 32 + g * 8];
                B1.h[0] = *(const v8h*)(wp);
                B1.h[1] = *(const v8h*)(wp + 16);
            }
            acc[0][0] = wmma_f16(A0, B0, acc[0][0]);
            acc[0][1] = wmma_f16(A0, B1, acc[0][1]);
            acc[1][0] = wmma_f16(A1, B0, acc[1][0]);
            acc[1][1] = wmma_f16(A1, B1, acc[1][1]);
        }
    }

    // ---- write back: lane holds D[m=r+8g][n=ln] of tile (mi, ni)
    const int h = h0 + mrow;
    #pragma unroll
    for (int mi = 0; mi < 2; ++mi) {
        #pragma unroll
        for (int ni = 0; ni < 2; ++ni) {
            const int co    = co0 + nt2 * 32 + ni * 16 + ln;
            const int wbase = mi * 16 + 8 * g;
            if (mode == 0) {
                #pragma unroll
                for (int r = 0; r < 8; ++r) {
                    int w = wbase + r;
                    out_h[((size_t)(h * WW + w) * NIMG + nimg) * CCH + co] =
                        (_Float16)acc[mi][ni][r];
                }
            } else {
                size_t base = (((size_t)nimg * CCH + co) * HH + h) * WW + wbase;
                #pragma unroll
                for (int r = 0; r < 8; ++r)
                    out_f[base + r] = resid[base + r] + acc[mi][ni][r];
            }
        }
    }
}

// ---------------------------------------------------------------------------
// Per-pixel cross-instance attention:
//   att = softmax_j( Q Kt / sqrt(C) ),  virt = att V
// q/k/v layout: [hw][n][c] f16.  Output: virt NCHW f32.
// ---------------------------------------------------------------------------
__global__ __launch_bounds__(256) void attention_wmma_kernel(
    const _Float16* __restrict__ q,
    const _Float16* __restrict__ k,
    const _Float16* __restrict__ v,
    float* __restrict__ virt)
{
    __shared__ float    attS[32 * 32];      // logits / exp
    __shared__ _Float16 attH[32 * 32];      // softmax result f16
    __shared__ _Float16 vt[CCH * 32];       // V transposed: [c][j]  32 KB

    const int tid  = threadIdx.x;
    const int wave = tid >> 5;
    const int lane = tid & 31;
    const int g    = lane >> 4;
    const int ln   = lane & 15;
    const int hw   = blockIdx.x;

    const _Float16* qb = q + (size_t)hw * NIMG * CCH;   // [i][c]
    const _Float16* kb = k + (size_t)hw * NIMG * CCH;   // [j][c]
    const _Float16* vb = v + (size_t)hw * NIMG * CCH;   // [j][c]

    // ---- phase 1: waves 0-3 compute att logits; waves 4-7 transpose V
    if (wave < 4) {
        const int mt = wave >> 1;           // i-subtile
        const int nt = wave & 1;            // j-subtile
        v8f acc = {};
        for (int cc = 0; cc < CCH / 32; ++cc) {
            Frag A, B;
            const _Float16* qp = &qb[(size_t)(mt * 16 + ln) * CCH + cc * 32];
            A.h[0] = *(const v8h*)(qp + g * 8);
            A.h[1] = *(const v8h*)(qp + 16 + g * 8);
            const _Float16* kp = &kb[(size_t)(nt * 16 + ln) * CCH + cc * 32];
            B.h[0] = *(const v8h*)(kp + g * 8);
            B.h[1] = *(const v8h*)(kp + 16 + g * 8);
            acc = wmma_f16(A, B, acc);
        }
        const float scale = 0.04419417382415922f;       // 1/sqrt(512)
        #pragma unroll
        for (int r = 0; r < 8; ++r)
            attS[(mt * 16 + 8 * g + r) * 32 + nt * 16 + ln] = acc[r] * scale;
    } else {
        for (int e = tid - 128; e < CCH * 32; e += 128) {
            int c = e & 511;
            int j = e >> 9;
            vt[c * 32 + j] = vb[(size_t)j * CCH + c];
        }
    }
    __syncthreads();

    // ---- softmax over j, one thread per row i
    if (tid < 32) {
        float mx = -3.4e38f;
        for (int j = 0; j < 32; ++j) mx = fmaxf(mx, attS[tid * 32 + j]);
        float s = 0.0f;
        for (int j = 0; j < 32; ++j) {
            float e = __expf(attS[tid * 32 + j] - mx);
            attS[tid * 32 + j] = e;
            s += e;
        }
        float inv = 1.0f / s;
        for (int j = 0; j < 32; ++j)
            attH[tid * 32 + j] = (_Float16)(attS[tid * 32 + j] * inv);
    }
    __syncthreads();

    // ---- phase 2: virt[i][c] = att[i][j] V[j][c]  (M=32,N=512,K=32)
    const int mt2 = wave >> 2;              // i-subtile for this wave
    Frag A;
    A.h[0] = *(const v8h*)&attH[(mt2 * 16 + ln) * 32 + g * 8];
    A.h[1] = *(const v8h*)&attH[(mt2 * 16 + ln) * 32 + 16 + g * 8];
    #pragma unroll
    for (int s = 0; s < 8; ++s) {
        const int ct = (wave & 3) + 4 * s;  // c-subtile 0..31
        Frag B;
        const _Float16* vp = &vt[(ct * 16 + ln) * 32 + g * 8];
        B.h[0] = *(const v8h*)(vp);
        B.h[1] = *(const v8h*)(vp + 16);
        v8f acc = {};
        acc = wmma_f16(A, B, acc);
        const int c = ct * 16 + ln;
        #pragma unroll
        for (int r = 0; r < 8; ++r) {
            int i = mt2 * 16 + 8 * g + r;
            virt[((size_t)i * CCH + c) * HW + hw] = acc[r];
        }
    }
}

// ---------------------------------------------------------------------------
// GroupNorm(1, C) stats per image over (C,H,W)
// ---------------------------------------------------------------------------
__global__ __launch_bounds__(256) void gn_stats_kernel(const float* __restrict__ virt,
                                                       float* __restrict__ stats) {
    __shared__ float rs[256], rs2[256];
    const int n = blockIdx.x;
    const int tid = threadIdx.x;
    const size_t per = (size_t)CCH * HW;
    const float* p = virt + (size_t)n * per;
    float s = 0.0f, s2 = 0.0f;
    for (size_t i = tid; i < per; i += 256) {
        float val = p[i];
        s += val;
        s2 += val * val;
    }
    rs[tid] = s; rs2[tid] = s2;
    __syncthreads();
    for (int off = 128; off > 0; off >>= 1) {
        if (tid < off) { rs[tid] += rs[tid + off]; rs2[tid] += rs2[tid + off]; }
        __syncthreads();
    }
    if (tid == 0) {
        float inv_n = 1.0f / (float)per;
        float mean = rs[0] * inv_n;
        float var = rs2[0] * inv_n - mean * mean;
        stats[n * 2 + 0] = mean;
        stats[n * 2 + 1] = rsqrtf(var + 1e-5f);
    }
}

// ---------------------------------------------------------------------------
// Normalize + affine + ReLU, in place
// ---------------------------------------------------------------------------
__global__ __launch_bounds__(256) void gn_apply_kernel(float* __restrict__ virt,
                                                       const float* __restrict__ stats,
                                                       const float* __restrict__ gamma,
                                                       const float* __restrict__ beta) {
    size_t idx = (size_t)blockIdx.x * 256 + threadIdx.x;
    if (idx >= (size_t)NIMG * CCH * HW) return;
    int n = (int)(idx >> 19);               // / (512*1024)
    int c = (int)((idx >> 10) & 511);
    float mean = stats[n * 2 + 0];
    float rstd = stats[n * 2 + 1];
    float y = (virt[idx] - mean) * rstd * gamma[c] + beta[c];
    virt[idx] = fmaxf(y, 0.0f);
}

// ---------------------------------------------------------------------------
extern "C" void kernel_launch(void* const* d_in, const int* in_sizes, int n_in,
                              void* d_out, int out_size, void* d_ws, size_t ws_size,
                              hipStream_t stream) {
    const float* x     = (const float*)d_in[0];
    const float* w_q   = (const float*)d_in[1];
    const float* w_k   = (const float*)d_in[2];
    const float* w_v   = (const float*)d_in[3];
    const float* w_o   = (const float*)d_in[4];
    const float* gamma = (const float*)d_in[5];
    const float* beta  = (const float*)d_in[6];
    float* out = (float*)d_out;

    const size_t WB  = (size_t)9 * CCH * CCH * sizeof(_Float16);   // 4.72 MB
    const size_t QKB = (size_t)HW * NIMG * CCH * sizeof(_Float16); // 33.6 MB
    const size_t VFB = (size_t)NIMG * CCH * HW * sizeof(float);    // 67.1 MB

    char* p = (char*)d_ws;
    _Float16* wqh = (_Float16*)p; p += WB;
    _Float16* wkh = (_Float16*)p; p += WB;
    _Float16* wvh = (_Float16*)p; p += WB;
    _Float16* woh = (_Float16*)p; p += WB;
    _Float16* qh  = (_Float16*)p; p += QKB;
    _Float16* kh  = (_Float16*)p; p += QKB;
    _Float16* vh  = (_Float16*)p; p += QKB;
    float*    virt = (float*)p;   p += VFB;
    float*    stats = (float*)p;

    const int wblocks = (9 * CCH * CCH + 255) / 256;
    repack_w_kernel<<<wblocks, 256, 0, stream>>>(w_q, wqh);
    repack_w_kernel<<<wblocks, 256, 0, stream>>>(w_k, wkh);
    repack_w_kernel<<<wblocks, 256, 0, stream>>>(w_v, wvh);
    repack_w_kernel<<<wblocks, 256, 0, stream>>>(w_o, woh);

    dim3 cgrid(NIMG * (HH / 4), CCH / 64);   // 256 x 8 blocks
    conv3x3_wmma_kernel<<<cgrid, 256, 0, stream>>>(x, wqh, qh, nullptr, nullptr, 0);
    conv3x3_wmma_kernel<<<cgrid, 256, 0, stream>>>(x, wkh, kh, nullptr, nullptr, 0);
    conv3x3_wmma_kernel<<<cgrid, 256, 0, stream>>>(x, wvh, vh, nullptr, nullptr, 0);

    attention_wmma_kernel<<<HW, 256, 0, stream>>>(qh, kh, vh, virt);

    gn_stats_kernel<<<NIMG, 256, 0, stream>>>(virt, stats);
    size_t tot = (size_t)NIMG * CCH * HW;
    gn_apply_kernel<<<(unsigned)((tot + 255) / 256), 256, 0, stream>>>(virt, stats, gamma, beta);

    conv3x3_wmma_kernel<<<cgrid, 256, 0, stream>>>(virt, woh, nullptr, out, x, 1);
}
